// TransformerBlock_89575837925998
// MI455X (gfx1250) — matrix operand order repaired and checked by reading, not compiled, not run on hardware
//
#include <hip/hip_runtime.h>
#include <hip/hip_bf16.h>
#include <math.h>

typedef __attribute__((ext_vector_type(16))) __bf16 v16bf;
typedef __attribute__((ext_vector_type(8)))  __bf16 v8bf;
typedef __attribute__((ext_vector_type(8)))  float  v8f;
typedef __attribute__((ext_vector_type(4)))  unsigned int v4u;
typedef __attribute__((ext_vector_type(8)))  int v8i;
typedef __attribute__((ext_vector_type(4)))  int v4i;

#define N_TOK 4096
#define FEAT  256
#define HEADS 8
#define KQDIM 64
#define HIDD  1024
#define DVDIM 32

__device__ __forceinline__ unsigned short f2bf(float f) {
  union { __bf16 h; unsigned short u; } c; c.h = (__bf16)f; return c.u;
}

// ---- TDM: async 2D tile load global -> LDS (bf16 elements, data_size = 2B) ----
// tensor viewed as [tensor_d1 rows][stride0 elems], tile = tile_d1 rows x tile_d0 elems,
// written contiguously (row-major) at lds_off. Tracked by TENSORcnt.
__device__ __forceinline__ void tdm_load_2d(unsigned lds_off, const void* gaddr,
                                            unsigned tensor_d0, unsigned tensor_d1,
                                            unsigned tile_d0, unsigned tile_d1,
                                            unsigned stride0) {
  unsigned long long ga = (unsigned long long)gaddr;
  v4u g0;
  g0.x = 1u;                                                  // count=1, user-mode descriptor
  g0.y = lds_off;                                             // lds_addr [63:32]
  g0.z = (unsigned)(ga & 0xFFFFFFFFu);                        // global_addr[31:0]
  g0.w = (unsigned)((ga >> 32) & 0x1FFFFFFu) | (2u << 30);    // global_addr[56:32] | type=2
  v8i g1;
  g1[0] = (int)(1u << 16);                                    // workgroup_mask=0, data_size=1 (2B)
  g1[1] = (int)((tensor_d0 & 0xFFFFu) << 16);                 // tensor_dim0[15:0]
  g1[2] = (int)((tensor_d0 >> 16) | ((tensor_d1 & 0xFFFFu) << 16));
  g1[3] = (int)((tensor_d1 >> 16) | (tile_d0 << 16));         // tensor_dim1[31:16] | tile_dim0
  g1[4] = (int)(tile_d1 & 0xFFFFu);                           // tile_dim1 (tile_dim2 = 0)
  g1[5] = (int)stride0;                                       // tensor_dim0_stride[31:0]
  g1[6] = 0;                                                  // stride0[47:32]=0, stride1[15:0]=0
  g1[7] = 0;
  v4i gz4 = {0, 0, 0, 0};                                     // 2D: groups 2/3 unused
  v8i gz8 = {0, 0, 0, 0, 0, 0, 0, 0};
  __builtin_amdgcn_tensor_load_to_lds(g0, g1, gz4, gz4, gz8, 0);
}

__device__ __forceinline__ unsigned lds_offset(const void* p) {
  // generic pointer to LDS: aperture in [63:32], byte offset in [31:0]
  return (unsigned)(unsigned long long)p;
}

// ---------------- pack / convert kernels ----------------
__global__ __launch_bounds__(256)
void cvt_f32_bf16(const float* __restrict__ in, unsigned short* __restrict__ out, int n) {
  int i = blockIdx.x * 256 + threadIdx.x;
  if (i < n) out[i] = f2bf(in[i]);
}

// in layout [H][IN][OUT] (f32) -> out layout [(h*OUT+o)][IN] (bf16, "Bt": row = out col, contiguous K)
__global__ __launch_bounds__(256)
void pack_weight(const float* __restrict__ in, unsigned short* __restrict__ out,
                 int H, int IN, int OUT) {
  int i = blockIdx.x * 256 + threadIdx.x;
  int total = H * IN * OUT;
  if (i >= total) return;
  int o = i % OUT;
  int t = i / OUT;
  int f = t % IN;
  int hh = t / IN;
  out[(size_t)(hh * OUT + o) * IN + f] = f2bf(in[i]);
}

// ---------------- generic bf16 WMMA GEMM ----------------
// C[M,Ncols] = A[M,K](bf16) * Bt[Ncols,K]^T (bf16) + bias, then (*scale, optional SiLU)
// mode: 0 = store f32 row-major, 1 = store bf16 row-major, 2 = store bf16 transposed [n*ldT+row]
__global__ __launch_bounds__(256)
void gemm_bf16(const unsigned short* __restrict__ Au,
               const unsigned short* __restrict__ Btu,
               const float* __restrict__ bias,
               float* __restrict__ outF,
               unsigned short* __restrict__ outB,
               int Ncols, int K, float scale, int act, int mode, int ldT) {
  const __bf16* A  = (const __bf16*)Au;
  const __bf16* Bt = (const __bf16*)Btu;
  const int lane = threadIdx.x & 31;
  const int wave = threadIdx.x >> 5;
  const int ln   = lane & 15;
  const int hlf  = lane >> 4;
  const int rowBase = blockIdx.y * 64 + (wave & 3) * 16;
  const int colBase = blockIdx.x * 128 + (wave >> 2) * 64;

  v8f acc[4] = {};
  const __bf16* aRow = A + (size_t)(rowBase + ln) * K;
  for (int kk = 0; kk < K; kk += 32) {
    union { v16bf v; v8bf p8[2]; } af;
    af.p8[0] = *(const v8bf*)(aRow + kk + 16 * hlf);
    af.p8[1] = *(const v8bf*)(aRow + kk + 16 * hlf + 8);
#pragma unroll
    for (int c = 0; c < 4; ++c) {
      const __bf16* bp = Bt + (size_t)(colBase + 16 * c + ln) * K + kk + 16 * hlf;
      v16bf bf = *(const v16bf*)bp;
      acc[c] = __builtin_amdgcn_wmma_f32_16x16x32_bf16(false, af.v, false, bf, (short)0,
                                                       acc[c], false, false);
    }
  }
#pragma unroll
  for (int c = 0; c < 4; ++c) {
    int n = colBase + 16 * c + ln;
    float bv = bias[n];
#pragma unroll
    for (int r = 0; r < 8; ++r) {
      int row = rowBase + r + 8 * hlf;
      float v = (acc[c][r] + bv) * scale;
      if (act == 1) v = v / (1.0f + __expf(-v));
      if (mode == 0)      outF[(size_t)row * Ncols + n] = v;
      else if (mode == 1) outB[(size_t)row * Ncols + n] = f2bf(v);
      else                outB[(size_t)n * ldT + row]   = f2bf(v);
    }
  }
}

// ---------------- flash attention with TDM double-buffered K/V tiles ----------------
// Qb, Kb: [N_TOK][HEADS*KQDIM] bf16 row-major (Q pre-scaled by 1/sqrt(FEAT))
// Vt:     [HEADS*DVDIM][N_TOK] bf16 (transposed)
// y:      [N_TOK][FEAT] f32; head hd fills cols [hd*DVDIM, (hd+1)*DVDIM)
// block = 4 waves, all on the same head: K/V tiles are shared -> TDM stages them in LDS.
__global__ __launch_bounds__(128)
void attn_flash(const unsigned short* __restrict__ Qu,
                const unsigned short* __restrict__ Ku,
                const unsigned short* __restrict__ Vu,
                float* __restrict__ y) {
  __shared__ __align__(16) __bf16 KtLds[2][32 * 64];   // 32 keys x 64 kq (4KB each)
  __shared__ __align__(16) __bf16 VtLds[2][32 * 32];   // 32 dv rows x 32 keys (2KB each)
  __shared__ __align__(32) __bf16 Pbuf[4][16 * 32];    // per-wave P scratch
  const __bf16* Qb = (const __bf16*)Qu;
  const __bf16* Kb = (const __bf16*)Ku;
  const __bf16* Vt = (const __bf16*)Vu;
  const int lane = threadIdx.x & 31;
  const int wave = threadIdx.x >> 5;
  const int ln  = lane & 15;
  const int hlf = lane >> 4;
  const int hd  = blockIdx.y;
  const int q0  = (blockIdx.x * 4 + wave) * 16;
  __bf16* P = Pbuf[wave];

  union F16 { v16bf v; v8bf p8[2]; };
  F16 qa[2];
  const __bf16* qrow = Qb + (size_t)(q0 + ln) * (HEADS * KQDIM) + hd * KQDIM;
#pragma unroll
  for (int c = 0; c < 2; ++c) {
    qa[c].p8[0] = *(const v8bf*)(qrow + 32 * c + 16 * hlf);
    qa[c].p8[1] = *(const v8bf*)(qrow + 32 * c + 16 * hlf + 8);
  }
  float m[8], l[8];
  v8f o0 = {}, o1 = {};
#pragma unroll
  for (int r = 0; r < 8; ++r) { m[r] = -__builtin_inff(); l[r] = 0.0f; }

  // prologue: stage tile 0 into buffer 0
  if (wave == 0) {
    tdm_load_2d(lds_offset(&KtLds[0][0]), Kb + (size_t)0 * (HEADS * KQDIM) + hd * KQDIM,
                HEADS * KQDIM, N_TOK, KQDIM, 32, HEADS * KQDIM);
    tdm_load_2d(lds_offset(&VtLds[0][0]), Vt + (size_t)(hd * DVDIM) * N_TOK + 0,
                N_TOK, HEADS * DVDIM, 32, DVDIM, N_TOK);
    __builtin_amdgcn_s_wait_tensorcnt(0);
  }
  __syncthreads();

  for (int kt = 0; kt < N_TOK / 32; ++kt) {
    const int key0 = kt * 32;
    const int buf  = kt & 1;
    // prefetch next tile into the other buffer (DMA overlaps with compute below)
    if (wave == 0 && key0 + 32 < N_TOK) {
      const int nk = key0 + 32;
      tdm_load_2d(lds_offset(&KtLds[buf ^ 1][0]),
                  Kb + (size_t)nk * (HEADS * KQDIM) + hd * KQDIM,
                  HEADS * KQDIM, N_TOK, KQDIM, 32, HEADS * KQDIM);
      tdm_load_2d(lds_offset(&VtLds[buf ^ 1][0]),
                  Vt + (size_t)(hd * DVDIM) * N_TOK + nk,
                  N_TOK, HEADS * DVDIM, 32, DVDIM, N_TOK);
    }
    const __bf16* Kt = KtLds[buf];
    const __bf16* Vl = VtLds[buf];

    v8f s0 = {}, s1 = {};
#pragma unroll
    for (int c = 0; c < 2; ++c) {
      v16bf b0 = *(const v16bf*)(Kt + (size_t)ln * KQDIM + 32 * c + 16 * hlf);
      v16bf b1 = *(const v16bf*)(Kt + (size_t)(16 + ln) * KQDIM + 32 * c + 16 * hlf);
      s0 = __builtin_amdgcn_wmma_f32_16x16x32_bf16(false, qa[c].v, false, b0, (short)0, s0, false, false);
      s1 = __builtin_amdgcn_wmma_f32_16x16x32_bf16(false, qa[c].v, false, b1, (short)0, s1, false, false);
    }
    // ---- online softmax (rows live across 16 lanes of each half) ----
    float t[8], alpha[8], rs[8];
#pragma unroll
    for (int r = 0; r < 8; ++r) t[r] = fmaxf(s0[r], s1[r]);
    for (int msk = 1; msk < 16; msk <<= 1) {
#pragma unroll
      for (int r = 0; r < 8; ++r) t[r] = fmaxf(t[r], __shfl_xor(t[r], msk, 16));
    }
#pragma unroll
    for (int r = 0; r < 8; ++r) {
      float mn = fmaxf(m[r], t[r]);
      alpha[r] = __expf(m[r] - mn);
      m[r] = mn;
      float p0 = __expf(s0[r] - mn);
      float p1 = __expf(s1[r] - mn);
      s0[r] = p0; s1[r] = p1;
      rs[r] = p0 + p1;
    }
    for (int msk = 1; msk < 16; msk <<= 1) {
#pragma unroll
      for (int r = 0; r < 8; ++r) rs[r] += __shfl_xor(rs[r], msk, 16);
    }
#pragma unroll
    for (int r = 0; r < 8; ++r) {
      l[r] = l[r] * alpha[r] + rs[r];
      o0[r] = o0[r] * alpha[r];
      o1[r] = o1[r] * alpha[r];
      int row = r + 8 * hlf;
      P[row * 32 + ln]      = (__bf16)s0[r];
      P[row * 32 + 16 + ln] = (__bf16)s1[r];
    }
    __syncthreads();
    F16 pa;
    pa.p8[0] = *(const v8bf*)(P + ln * 32 + 16 * hlf);
    pa.p8[1] = *(const v8bf*)(P + ln * 32 + 16 * hlf + 8);
    v16bf vb0 = *(const v16bf*)(Vl + (size_t)ln * 32 + 16 * hlf);
    v16bf vb1 = *(const v16bf*)(Vl + (size_t)(16 + ln) * 32 + 16 * hlf);
    o0 = __builtin_amdgcn_wmma_f32_16x16x32_bf16(false, pa.v, false, vb0, (short)0, o0, false, false);
    o1 = __builtin_amdgcn_wmma_f32_16x16x32_bf16(false, pa.v, false, vb1, (short)0, o1, false, false);
    // next buffer must be resident before the next iteration reads it
    if (wave == 0) __builtin_amdgcn_s_wait_tensorcnt(0);
    __syncthreads();
  }
#pragma unroll
  for (int r = 0; r < 8; ++r) {
    float inv = 1.0f / l[r];
    int row = q0 + r + 8 * hlf;
    y[(size_t)row * FEAT + hd * DVDIM + ln]      = o0[r] * inv;
    y[(size_t)row * FEAT + hd * DVDIM + 16 + ln] = o1[r] * inv;
  }
}

// ---------------- fused add + RMSNorm ----------------
__global__ __launch_bounds__(256)
void add_rmsnorm(const float* __restrict__ a, const float* __restrict__ b,
                 const float* __restrict__ g, float* __restrict__ outF,
                 unsigned short* __restrict__ outB) {
  __shared__ float red[8];
  const int row = blockIdx.x;
  const int tid = threadIdx.x;
  float v = a[(size_t)row * FEAT + tid] + b[(size_t)row * FEAT + tid];
  float ss = v * v;
  for (int msk = 16; msk >= 1; msk >>= 1) ss += __shfl_xor(ss, msk, 32);
  if ((tid & 31) == 0) red[tid >> 5] = ss;
  __syncthreads();
  float tot = 0.0f;
#pragma unroll
  for (int i = 0; i < 8; ++i) tot += red[i];
  float rms = rsqrtf(tot * (1.0f / FEAT) + 1.1920929e-07f);  // finfo(f32).eps
  float z = v * rms * g[tid];
  outF[(size_t)row * FEAT + tid] = z;
  if (outB) outB[(size_t)row * FEAT + tid] = f2bf(z);
}

// ---------------- launcher ----------------
extern "C" void kernel_launch(void* const* d_in, const int* in_sizes, int n_in,
                              void* d_out, int out_size, void* d_ws, size_t ws_size,
                              hipStream_t stream) {
  (void)in_sizes; (void)n_in; (void)out_size; (void)ws_size;
  const float* x  = (const float*)d_in[0];
  const float* Wq = (const float*)d_in[1];
  const float* bq = (const float*)d_in[2];
  const float* Wk = (const float*)d_in[3];
  const float* bk = (const float*)d_in[4];
  const float* Wv = (const float*)d_in[5];
  const float* bv = (const float*)d_in[6];
  const float* g1 = (const float*)d_in[7];
  const float* g2 = (const float*)d_in[8];
  const float* W1 = (const float*)d_in[9];
  const float* b1 = (const float*)d_in[10];
  const float* W2 = (const float*)d_in[11];
  const float* b2 = (const float*)d_in[12];
  const float* W3 = (const float*)d_in[13];
  const float* b3 = (const float*)d_in[14];
  float* out = (float*)d_out;

  char* w = (char*)d_ws;
  size_t off = 0;
  auto carve = [&](size_t bytes) -> void* {
    void* p = w + off;
    off += (bytes + 255) & ~(size_t)255;
    return p;
  };
  unsigned short* xb  = (unsigned short*)carve((size_t)N_TOK * FEAT * 2);
  unsigned short* Wqt = (unsigned short*)carve((size_t)HEADS * KQDIM * FEAT * 2);
  unsigned short* Wkt = (unsigned short*)carve((size_t)HEADS * KQDIM * FEAT * 2);
  unsigned short* Wvt = (unsigned short*)carve((size_t)HEADS * DVDIM * FEAT * 2);
  unsigned short* W1t = (unsigned short*)carve((size_t)FEAT * HIDD * 2);
  unsigned short* W2t = (unsigned short*)carve((size_t)HIDD * HIDD * 2);
  unsigned short* W3t = (unsigned short*)carve((size_t)HIDD * FEAT * 2);
  unsigned short* Qb  = (unsigned short*)carve((size_t)N_TOK * HEADS * KQDIM * 2);
  unsigned short* Kb  = (unsigned short*)carve((size_t)N_TOK * HEADS * KQDIM * 2);
  unsigned short* Vt  = (unsigned short*)carve((size_t)HEADS * DVDIM * N_TOK * 2);
  float*          yv  = (float*)carve((size_t)N_TOK * FEAT * 4);
  float*          zf  = (float*)carve((size_t)N_TOK * FEAT * 4);
  unsigned short* zb  = (unsigned short*)carve((size_t)N_TOK * FEAT * 2);
  unsigned short* h1  = (unsigned short*)carve((size_t)N_TOK * HIDD * 2);
  unsigned short* h2  = (unsigned short*)carve((size_t)N_TOK * HIDD * 2);
  float*          pb  = (float*)carve((size_t)N_TOK * FEAT * 4);

  // pack inputs/weights to bf16 (transposed weights: row = output column)
  cvt_f32_bf16<<<(N_TOK * FEAT + 255) / 256, 256, 0, stream>>>(x, xb, N_TOK * FEAT);
  pack_weight<<<(HEADS * FEAT * KQDIM + 255) / 256, 256, 0, stream>>>(Wq, Wqt, HEADS, FEAT, KQDIM);
  pack_weight<<<(HEADS * FEAT * KQDIM + 255) / 256, 256, 0, stream>>>(Wk, Wkt, HEADS, FEAT, KQDIM);
  pack_weight<<<(HEADS * FEAT * DVDIM + 255) / 256, 256, 0, stream>>>(Wv, Wvt, HEADS, FEAT, DVDIM);
  pack_weight<<<(FEAT * HIDD + 255) / 256, 256, 0, stream>>>(W1, W1t, 1, FEAT, HIDD);
  pack_weight<<<(HIDD * HIDD + 255) / 256, 256, 0, stream>>>(W2, W2t, 1, HIDD, HIDD);
  pack_weight<<<(HIDD * FEAT + 255) / 256, 256, 0, stream>>>(W3, W3t, 1, HIDD, FEAT);

  // Q (scaled by 1/sqrt(FEAT)=1/16), K, V projections
  gemm_bf16<<<dim3((HEADS * KQDIM) / 128, N_TOK / 64), 256, 0, stream>>>(
      xb, Wqt, bq, nullptr, Qb, HEADS * KQDIM, FEAT, 1.0f / 16.0f, 0, 1, 0);
  gemm_bf16<<<dim3((HEADS * KQDIM) / 128, N_TOK / 64), 256, 0, stream>>>(
      xb, Wkt, bk, nullptr, Kb, HEADS * KQDIM, FEAT, 1.0f, 0, 1, 0);
  gemm_bf16<<<dim3((HEADS * DVDIM) / 128, N_TOK / 64), 256, 0, stream>>>(
      xb, Wvt, bv, nullptr, Vt, HEADS * DVDIM, FEAT, 1.0f, 0, 2, N_TOK);

  // attention: grid = (query blocks of 64 rows, heads), 4 waves/block, 16 rows/wave
  attn_flash<<<dim3(N_TOK / 64, HEADS), 128, 0, stream>>>(Qb, Kb, Vt, yv);

  // z = rmsnorm(x + y, g1)  (also bf16 copy for MLP)
  add_rmsnorm<<<N_TOK, 256, 0, stream>>>(x, yv, g1, zf, zb);

  // SiLU MLP
  gemm_bf16<<<dim3(HIDD / 128, N_TOK / 64), 256, 0, stream>>>(
      zb, W1t, b1, nullptr, h1, HIDD, FEAT, 1.0f, 1, 1, 0);
  gemm_bf16<<<dim3(HIDD / 128, N_TOK / 64), 256, 0, stream>>>(
      h1, W2t, b2, nullptr, h2, HIDD, HIDD, 1.0f, 1, 1, 0);
  gemm_bf16<<<dim3(FEAT / 128, N_TOK / 64), 256, 0, stream>>>(
      h2, W3t, b3, pb, nullptr, FEAT, HIDD, 1.0f, 0, 0, 0);

  // out = rmsnorm(p + z, g2)
  add_rmsnorm<<<N_TOK, 256, 0, stream>>>(pb, zf, g2, out, nullptr);
}
